// GINModel_91010357002576
// MI455X (gfx1250) — matrix operand order repaired and checked by reading, not compiled, not run on hardware
//
#include <hip/hip_runtime.h>
#include <hip/hip_bf16.h>

// ---------------------------------------------------------------------------
// GIN (2x GINEConv) for MI455X / gfx1250.
//   conv(x): y = x*W1a + b1 (per node, WMMA), then per message
//            h1 = relu(y[src] + ea*W1b); out[dst] += h1*W2 + b2 (WMMA + atomics)
// bf16 WMMA (v_wmma_f32_16x16x32_bf16), fp32 accumulate.
// y tables (51MB) are L2-resident (192MB L2); scatter atomics resolve in L2.
// Block-cooperative tile loop amortizes LDS weight staging without the
// register-pressure blowup of a free-running wave loop (round-2 spilled).
// ---------------------------------------------------------------------------

typedef __attribute__((ext_vector_type(16))) __bf16 v16bf;
typedef __attribute__((ext_vector_type(8)))  float  v8f;

__device__ __forceinline__ v8f wmma_bf16(v16bf a, v16bf b, v8f c) {
  // (neg_a, A, neg_b, B, c_mod, C, reuse_a, reuse_b)
  return __builtin_amdgcn_wmma_f32_16x16x32_bf16(false, a, false, b, (short)0, c,
                                                 false, false);
}

// ---------------------------------------------------------------------------
// zero-fill (float4 stores)
// ---------------------------------------------------------------------------
__global__ void zero_f4_kernel(float4* __restrict__ p, long n4) {
  long i = (long)blockIdx.x * blockDim.x + threadIdx.x;
  long stride = (long)gridDim.x * blockDim.x;
  float4 z = make_float4(0.f, 0.f, 0.f, 0.f);
  for (; i < n4; i += stride) p[i] = z;
}

// ---------------------------------------------------------------------------
// y[node][0..127] = (relu?)(in[node][0..K-1]) * W[0..K-1][0..127] + bias
// One wave = one 16-node tile per loop iteration. 8 waves / block.
// ---------------------------------------------------------------------------
template <int K, int RELU_IN>
__global__ void node_linear_kernel(const float* __restrict__ in,
                                   const float* __restrict__ W,     // [K][128]
                                   const float* __restrict__ bias,  // [128]
                                   float* __restrict__ out,         // [N][128]
                                   int N, long nTiles) {
  constexpr int NKC = K / 32;
  __shared__ __align__(32) __bf16 wt[128 * K];   // W^T: [n][k], bf16
  __shared__ float bs[128];

  // Stage W^T (coalesced global reads) + bias, once per block.
  for (int idx = threadIdx.x; idx < 128 * K; idx += blockDim.x) {
    int n = idx & 127;
    int k = idx >> 7;
    wt[n * K + k] = (__bf16)W[k * 128 + n];
  }
  for (int c = threadIdx.x; c < 128; c += blockDim.x) bs[c] = bias[c];
  __syncthreads();

  const int lane = threadIdx.x & 31;
  const int wave = threadIdx.x >> 5;
  const int ml = lane & 15;   // row within 16-node tile
  const int hi = lane >> 4;   // K-half selector (A) / M-half selector (D)
  const long gwave = (long)blockIdx.x * 8 + wave;
  const long nwaves = (long)gridDim.x * 8;

  for (long tile = gwave; tile < nTiles; tile += nwaves) {
    const int base = (int)(tile * 16);

    // ---- Build A (16 x K, bf16) with k = 16*hi + s (contiguous) -----------
    int node = base + ml;
    int ns = node < N ? node : (N - 1);  // clamp loads; stores masked below
    const float* src = in + (long)ns * K;

    v16bf a[NKC];
#pragma unroll
    for (int kc = 0; kc < NKC; ++kc) {
#pragma unroll
      for (int g = 0; g < 2; ++g) {
        const int c0 = kc * 32 + hi * 16 + g * 8;
        float4 f0 = *reinterpret_cast<const float4*>(src + c0);
        float4 f1 = *reinterpret_cast<const float4*>(src + c0 + 4);
        float v[8] = {f0.x, f0.y, f0.z, f0.w, f1.x, f1.y, f1.z, f1.w};
#pragma unroll
        for (int j = 0; j < 8; ++j) {
          float t = v[j];
          if (RELU_IN) t = fmaxf(t, 0.f);
          a[kc][g * 8 + j] = (__bf16)t;
        }
      }
    }

    // ---- 16x128 output tile: 8 N-tiles x NKC K-chunks of WMMA -----------
#pragma unroll
    for (int nt = 0; nt < 8; ++nt) {
      v8f acc = {};
#pragma unroll
      for (int kc = 0; kc < NKC; ++kc) {
        const v16bf b = *reinterpret_cast<const v16bf*>(
            &wt[(nt * 16 + ml) * K + kc * 32 + hi * 16]);
        acc = wmma_bf16(a[kc], b, acc);
      }
      const int c = nt * 16 + ml;
      const float bv = bs[c];
#pragma unroll
      for (int j = 0; j < 8; ++j) {
        int row = base + hi * 8 + j;
        if (row < N) out[(long)row * 128 + c] = acc[j] + bv;
      }
    }
  }
}

// ---------------------------------------------------------------------------
// Per-message MLP + scatter-add.
// Messages 0..E-1 are edges; E..E+N-1 are self-loops (ea = 1,1).
//   h1 = relu(y[src] + ea0*Wb[0] + ea1*Wb[1])        (y has b1 folded in)
//   atomicAdd(out[dst], h1*W2 + b2)
// Block-cooperative loop: all 8 waves advance one 16-message tile per
// iteration; metadata lives in LDS (low register pressure, as in round 1).
// ---------------------------------------------------------------------------
__global__ void edge_mlp_scatter_kernel(const float* __restrict__ y,     // [N][128]
                                        const long long* __restrict__ srcI,
                                        const long long* __restrict__ dstI,
                                        const float* __restrict__ ea,    // [E][2]
                                        const float* __restrict__ Wb,    // [2][128]
                                        const float* __restrict__ W2,    // [128][128]
                                        const float* __restrict__ b2,    // [128]
                                        float* __restrict__ out,         // [N][128]
                                        int N, int E, long nTiles, int nIter) {
  __shared__ __align__(32) __bf16 wt2[128 * 128];  // W2^T bf16 (32KB)
  __shared__ float wb0[128], wb1[128], b2s[128];
  __shared__ int   ms[8][16], md[8][16];
  __shared__ float me0[8][16], me1[8][16];

  for (int idx = threadIdx.x; idx < 128 * 128; idx += blockDim.x) {
    int n = idx & 127;
    int k = idx >> 7;
    wt2[n * 128 + k] = (__bf16)W2[k * 128 + n];
  }
  for (int c = threadIdx.x; c < 128; c += blockDim.x) {
    wb0[c] = Wb[c];
    wb1[c] = Wb[128 + c];
    b2s[c] = b2[c];
  }
  __syncthreads();

  const int lane = threadIdx.x & 31;
  const int wave = threadIdx.x >> 5;
  const int ml = lane & 15;
  const int hi = lane >> 4;
  const long M = (long)E + (long)N;
  const long waveStride = (long)gridDim.x * 8;

  long tile = (long)blockIdx.x * 8 + wave;
  for (int it = 0; it < nIter; ++it, tile += waveStride) {
    // ---- metadata phase (lanes 0..15) ----------------------------------
    if (lane < 16) {
      long i = tile * 16 + lane;
      int s = 0, d = -1;
      float e0 = 1.f, e1 = 1.f;
      if (tile < nTiles && i < M) {
        if (i < (long)E) {
          s = (int)srcI[i];
          d = (int)dstI[i];
          e0 = ea[i * 2 + 0];
          e1 = ea[i * 2 + 1];
        } else {
          s = (int)(i - E);
          d = s;
        }
      }
      ms[wave][lane] = s;
      md[wave][lane] = d;
      me0[wave][lane] = e0;
      me1[wave][lane] = e1;

      // Prefetch next iteration's edge metadata (global_prefetch_b8) —
      // overlaps the index fetch with this iteration's 32-WMMA GEMM.
      long ipf = i + waveStride * 16;
      if (ipf < (long)E) {
        __builtin_prefetch(srcI + ipf, 0, 3);
        __builtin_prefetch(dstI + ipf, 0, 3);
        __builtin_prefetch(ea + ipf * 2, 0, 3);
      }
    }
    __syncthreads();

    if (tile < nTiles) {  // wave-uniform: WMMA runs with full EXEC
      // ---- h1 = relu(y[src] + ea*Wb), packed into the WMMA A layout -----
      const int s = ms[wave][ml];
      const float e0 = me0[wave][ml];
      const float e1 = me1[wave][ml];
      const float* ys = y + (long)s * 128;

      v16bf a[4];
#pragma unroll
      for (int kc = 0; kc < 4; ++kc) {
#pragma unroll
        for (int g = 0; g < 2; ++g) {
          const int c0 = kc * 32 + hi * 16 + g * 8;
          float4 f0 = *reinterpret_cast<const float4*>(ys + c0);
          float4 f1 = *reinterpret_cast<const float4*>(ys + c0 + 4);
          float v[8] = {f0.x, f0.y, f0.z, f0.w, f1.x, f1.y, f1.z, f1.w};
#pragma unroll
          for (int j = 0; j < 8; ++j) {
            float t = v[j] + e0 * wb0[c0 + j] + e1 * wb1[c0 + j];
            t = fmaxf(t, 0.f);
            a[kc][g * 8 + j] = (__bf16)t;
          }
        }
      }

      // ---- 16x128x128 GEMM (32 WMMAs) + bias + scatter-add --------------
#pragma unroll
      for (int nt = 0; nt < 8; ++nt) {
        v8f acc = {};
#pragma unroll
        for (int kc = 0; kc < 4; ++kc) {
          const v16bf b = *reinterpret_cast<const v16bf*>(
              &wt2[(nt * 16 + ml) * 128 + kc * 32 + hi * 16]);
          acc = wmma_bf16(a[kc], b, acc);
        }
        const int c = nt * 16 + ml;
        const float bv = b2s[c];
#pragma unroll
        for (int j = 0; j < 8; ++j) {
          const int d = md[wave][hi * 8 + j];
          if (d >= 0) atomicAdd(out + (long)d * 128 + c, acc[j] + bv);
        }
      }
    }
    __syncthreads();  // metadata arrays reused next iteration
  }
}

// ---------------------------------------------------------------------------
// Launch: y1 = x*W11a+b11 ; agg1 = scatter(edgeMLP1) ; y2 = relu(agg1)*W21a+b21 ;
//         out = scatter(edgeMLP2)
// ---------------------------------------------------------------------------
extern "C" void kernel_launch(void* const* d_in, const int* in_sizes, int n_in,
                              void* d_out, int out_size, void* d_ws, size_t ws_size,
                              hipStream_t stream) {
  (void)in_sizes; (void)n_in; (void)out_size; (void)ws_size;

  const float*      x   = (const float*)d_in[0];
  const long long*  ei  = (const long long*)d_in[1];  // [2][E] int64
  const float*      ea  = (const float*)d_in[2];      // [E][2]
  const float*      W11 = (const float*)d_in[3];      // [66][128]
  const float*      b11 = (const float*)d_in[4];
  const float*      W12 = (const float*)d_in[5];      // [128][128]
  const float*      b12 = (const float*)d_in[6];
  const float*      W21 = (const float*)d_in[7];      // [130][128]
  const float*      b21 = (const float*)d_in[8];
  const float*      W22 = (const float*)d_in[9];      // [128][128]
  const float*      b22 = (const float*)d_in[10];
  float*            out = (float*)d_out;

  const int N = 100000;
  const int E = 1600000;

  float* y   = (float*)d_ws;            // [N][128]  (y1, later reused as y2)
  float* agg = y + (size_t)N * 128;     // [N][128]  (conv1 aggregation)

  const long nodeTiles = (N + 15) / 16;       // 6250
  const long M         = (long)E + N;
  const long msgTiles  = (M + 15) / 16;       // 106250

  const int nodeBlocks = 512;   // 4096 waves, ~1.5 tiles each
  const int edgeBlocks = 2048;  // 16384 waves
  const int nIterEdge  = (int)((msgTiles + (long)edgeBlocks * 8 - 1) /
                               ((long)edgeBlocks * 8));  // 7
  const int zBlocks = 2048;
  const long n4 = (long)N * 128 / 4;

  // conv1
  node_linear_kernel<64, 0><<<nodeBlocks, 256, 0, stream>>>(x, W11, b11, y, N,
                                                            nodeTiles);
  zero_f4_kernel<<<zBlocks, 256, 0, stream>>>((float4*)agg, n4);
  edge_mlp_scatter_kernel<<<edgeBlocks, 256, 0, stream>>>(
      y, ei, ei + (size_t)E, ea, W11 + 64 * 128, W12, b12, agg, N, E, msgTiles,
      nIterEdge);

  // relu + conv2
  node_linear_kernel<128, 1><<<nodeBlocks, 256, 0, stream>>>(agg, W21, b21, y, N,
                                                             nodeTiles);
  zero_f4_kernel<<<zBlocks, 256, 0, stream>>>((float4*)out, n4);
  edge_mlp_scatter_kernel<<<edgeBlocks, 256, 0, stream>>>(
      y, ei, ei + (size_t)E, ea, W21 + 128 * 128, W22, b22, out, N, E, msgTiles,
      nIterEdge);
}
